// UNet_SCN_regressor_22419729285188
// MI455X (gfx1250) — matrix operand order repaired and checked by reading, not compiled, not run on hardware
//
#include <hip/hip_runtime.h>
#include <hip/hip_bf16.h>

// ---------------------------------------------------------------------------
// Types for CDNA5 WMMA (wave32): A/B are 16 x bf16 per lane, C/D are 8 x f32.
// ---------------------------------------------------------------------------
typedef __bf16 bf16_t;
typedef bf16_t v16bf __attribute__((ext_vector_type(16)));
typedef bf16_t v8bf  __attribute__((ext_vector_type(8)));
typedef float  v8f   __attribute__((ext_vector_type(8)));

#define WAVES_PER_BLOCK 8   // 256 threads = 8 wave32
#define NT 4                // N-tiles (16 couts each) per wave
#define MT 2                // M-tiles (16 voxels each) per wave -> B reuse x2
#define NB_STATS 64         // fixed partial-block count (deterministic reduce)

// ---------------------------------------------------------------------------
// Weight repack: w fp32 [Cout][Cin][27]  ->  bf16 [Npad][27][Cpad]
// (k = tap*Cpad + cin; zero padding for n>=Cout, c>=Cin).
// ---------------------------------------------------------------------------
__global__ __launch_bounds__(256)
void repack_w(const float* __restrict__ w, bf16_t* __restrict__ out,
              int Cout, int Cin, int Npad, int Cpad)
{
  const long i = (long)blockIdx.x * 256 + threadIdx.x;
  const long tot = (long)Npad * 27 * Cpad;
  if (i >= tot) return;
  const int c   = (int)(i % Cpad);
  const long t2 = i / Cpad;
  const int tap = (int)(t2 % 27);
  const int n   = (int)(t2 / 27);
  float v = 0.f;
  if (n < Cout && c < Cin) v = w[((long)n * Cin + c) * 27 + tap];
  out[i] = (bf16_t)v;
}

// import x: fp32 [3][M1] channel-first -> bf16 [M1][32] channel-last, zero-pad
__global__ __launch_bounds__(256)
void import_x(const float* __restrict__ x, bf16_t* __restrict__ out, int M)
{
  const long i = (long)blockIdx.x * 256 + threadIdx.x;
  if (i >= (long)M * 32) return;
  const int c = (int)(i & 31);
  const long m = i >> 5;
  out[i] = (bf16_t)((c < 3) ? x[(long)c * M + m] : 0.f);
}

// ---------------------------------------------------------------------------
// Implicit-GEMM 3D conv, activations bf16 CHANNEL-LAST [D][H][W][Cpad].
// GEMM: M = Do*Ho*Wo, N = Cout, K = 27*Cpad with k = tap*Cpad + cin.
// One wave computes a 32(M) x 64(N) tile: 2 A fragments x 4 B fragments,
// 8 v_wmma_f32_16x16x32_bf16 per K-chunk; each B fragment is reused for
// both M halves (1.5 loads per WMMA). A chunk is contiguous -> 2x b128.
// ---------------------------------------------------------------------------
__global__ __launch_bounds__(256)
void conv3d_wmma_cl(const bf16_t* __restrict__ in, const bf16_t* __restrict__ wB,
                    const float* __restrict__ bias,
                    bf16_t* __restrict__ out, float* __restrict__ out32,
                    int Cpad, int Cout, int ldo, int coff,
                    int Di, int Hi, int Wi,
                    int Do, int Ho, int Wo,
                    int sd, int sh, int sw,
                    int pd, int ph, int pw,
                    int ldil)
{
  const int lane = threadIdx.x & 31;
  const int wave = threadIdx.x >> 5;
  const int Mtot = Do * Ho * Wo;
  const int mtiles = (Mtot + (16 * MT - 1)) / (16 * MT);
  const int mtile = blockIdx.x * WAVES_PER_BLOCK + wave;
  if (mtile >= mtiles) return;                 // wave-uniform exit
  const int n0 = blockIdx.y * (16 * NT);
  const int m0 = mtile * (16 * MT);

  const int half = lane >> 4;                  // K-half selector
  const int r    = lane & 15;

  // Output voxels of this lane's two A rows (divisions happen once).
  const int HiWi = Hi * Wi;
  bool mv[MT];
  long arow0[MT];                              // spatial-independent precompute
  int odv[MT], ohv[MT], owv[MT];
#pragma unroll
  for (int mt = 0; mt < MT; ++mt) {
    const int mA = m0 + mt * 16 + r;
    mv[mt] = (mA < Mtot);
    int od = 0, oh = 0, ow = 0;
    if (mv[mt]) {
      od = mA / (Ho * Wo);
      int rem = mA - od * Ho * Wo;
      oh = rem / Wo;
      ow = rem - oh * Wo;
    }
    odv[mt] = od; ohv[mt] = oh; owv[mt] = ow;
    arow0[mt] = 0;
  }
  const int ccs = Cpad >> 5;                   // 32-chunks per tap

  // Per-N-tile B row base pointers (row = cout, 27*Cpad bf16 per row).
  const bf16_t* wrow[NT];
#pragma unroll
  for (int t = 0; t < NT; ++t)
    wrow[t] = wB + (long)(n0 + t * 16 + r) * (27L * Cpad) + half * 16;

  v8f z8 = (v8f){0.f,0.f,0.f,0.f,0.f,0.f,0.f,0.f};
  v8f a00 = z8, a01 = z8, a02 = z8, a03 = z8;  // M half 0, N tiles 0..3
  v8f a10 = z8, a11 = z8, a12 = z8, a13 = z8;  // M half 1, N tiles 0..3

  v8bf zero8;
#pragma unroll
  for (int e = 0; e < 8; ++e) zero8[e] = (bf16_t)0.f;

  int kd = 0, kh = 0, kw = 0;
  for (int tap = 0; tap < 27; ++tap) {
    // per-tap (per-lane) spatial offsets + validity for both M halves
    bool valid[MT];
#pragma unroll
    for (int mt = 0; mt < MT; ++mt) {
      const int ed = odv[mt] * sd + kd - pd;
      const int eh = ohv[mt] * sh + kh - ph;
      const int ew = owv[mt] * sw + kw - pw;
      bool v = mv[mt] && (ed >= 0) && (eh >= 0) && (ew >= 0);
      int idd = ed, ihh = eh, iww = ew;
      if (ldil == 2) {                         // transposed conv (input dil. 2)
        v = v && (((ed | eh | ew) & 1) == 0);
        idd = ed >> 1; ihh = eh >> 1; iww = ew >> 1;
      }
      v = v && (idd < Di) && (ihh < Hi) && (iww < Wi);
      valid[mt] = v;
      arow0[mt] = ((long)idd * HiWi + (long)ihh * Wi + iww) * Cpad + half * 16;
    }
    const long wtap = (long)tap * Cpad;

    for (int cc = 0; cc < ccs; ++cc) {
      const long coffk = (long)(cc << 5);
      // A fragments: elements 0-15 = K 0-15(+16*half)
      v8bf lo0 = zero8, hi0 = zero8, lo1 = zero8, hi1 = zero8;
      if (valid[0]) {
        const bf16_t* p = in + arow0[0] + coffk;
        lo0 = *(const v8bf*)(p);
        hi0 = *(const v8bf*)(p + 8);
      }
      if (valid[1]) {
        const bf16_t* p = in + arow0[1] + coffk;
        lo1 = *(const v8bf*)(p);
        hi1 = *(const v8bf*)(p + 8);
      }
      const v16bf af0 = __builtin_shufflevector(lo0, hi0,
          0,1,2,3,4,5,6,7,8,9,10,11,12,13,14,15);
      const v16bf af1 = __builtin_shufflevector(lo1, hi1,
          0,1,2,3,4,5,6,7,8,9,10,11,12,13,14,15);
      // B fragments: contiguous pre-packed bf16; each reused for both M halves
      const long boff = wtap + coffk;
      const v16bf b0 = *(const v16bf*)(wrow[0] + boff);
      const v16bf b1 = *(const v16bf*)(wrow[1] + boff);
      const v16bf b2 = *(const v16bf*)(wrow[2] + boff);
      const v16bf b3 = *(const v16bf*)(wrow[3] + boff);
      a00 = __builtin_amdgcn_wmma_f32_16x16x32_bf16(false, af0, false, b0,
                                                    (short)0, a00, false, false);
      a10 = __builtin_amdgcn_wmma_f32_16x16x32_bf16(false, af1, false, b0,
                                                    (short)0, a10, false, false);
      a01 = __builtin_amdgcn_wmma_f32_16x16x32_bf16(false, af0, false, b1,
                                                    (short)0, a01, false, false);
      a11 = __builtin_amdgcn_wmma_f32_16x16x32_bf16(false, af1, false, b1,
                                                    (short)0, a11, false, false);
      a02 = __builtin_amdgcn_wmma_f32_16x16x32_bf16(false, af0, false, b2,
                                                    (short)0, a02, false, false);
      a12 = __builtin_amdgcn_wmma_f32_16x16x32_bf16(false, af1, false, b2,
                                                    (short)0, a12, false, false);
      a03 = __builtin_amdgcn_wmma_f32_16x16x32_bf16(false, af0, false, b3,
                                                    (short)0, a03, false, false);
      a13 = __builtin_amdgcn_wmma_f32_16x16x32_bf16(false, af1, false, b3,
                                                    (short)0, a13, false, false);
    }
    if (++kw == 3) { kw = 0; if (++kh == 3) { kh = 0; ++kd; } }
  }

  // store: C/D layout: VGPR vv -> M = vv + 8*half, N = lane&15
  v8f accs[MT][NT] = {{a00, a01, a02, a03}, {a10, a11, a12, a13}};
#pragma unroll
  for (int mt = 0; mt < MT; ++mt) {
#pragma unroll
    for (int t = 0; t < NT; ++t) {
      const int n = n0 + t * 16 + r;
      if (n >= Cout) continue;
      const float bv = (bias != nullptr) ? bias[n] : 0.f;
      if (out32 != nullptr) {                  // final layer: fp32 channel-first
#pragma unroll
        for (int vv = 0; vv < 8; ++vv) {
          const int m = m0 + mt * 16 + vv + 8 * half;
          if (m < Mtot) out32[(long)n * Mtot + m] = accs[mt][t][vv] + bv;
        }
      } else {                                 // bf16 channel-last (into cat)
#pragma unroll
        for (int vv = 0; vv < 8; ++vv) {
          const int m = m0 + mt * 16 + vv + 8 * half;
          if (m < Mtot)
            out[(long)m * ldo + coff + n] = (bf16_t)(accs[mt][t][vv] + bv);
        }
      }
    }
  }
}

// ---------------------------------------------------------------------------
// BN stats over channel-last bf16, deterministic 2-stage reduction.
// Requires C in {32,64,128} (divides 256): thread t always sees channel
// t&(C-1), so per-channel partials reduce with a fixed LDS tree.
// ---------------------------------------------------------------------------
__global__ __launch_bounds__(256)
void bn_stats_cl_part(const bf16_t* __restrict__ x, int ldx, int xoff,
                      float* __restrict__ part, int C, int cshift, int M)
{
  __shared__ float ssum[256];
  __shared__ float ssq[256];
  const long total = (long)M << cshift;
  float s = 0.f, q = 0.f;
  for (long i = (long)blockIdx.x * 256 + threadIdx.x; i < total;
       i += (long)NB_STATS * 256) {
    const long m = i >> cshift;
    const int  c = (int)(i & (C - 1));
    const float v = (float)x[m * ldx + xoff + c];
    s += v; q += v * v;
  }
  ssum[threadIdx.x] = s; ssq[threadIdx.x] = q;
  __syncthreads();
  for (int off = 128; off >= C; off >>= 1) {
    if ((int)threadIdx.x < off) {
      ssum[threadIdx.x] += ssum[threadIdx.x + off];
      ssq[threadIdx.x]  += ssq[threadIdx.x + off];
    }
    __syncthreads();
  }
  if ((int)threadIdx.x < C) {
    part[blockIdx.x * 512 + threadIdx.x]       = ssum[threadIdx.x];
    part[blockIdx.x * 512 + 256 + threadIdx.x] = ssq[threadIdx.x];
  }
}

__global__ __launch_bounds__(256)
void bn_stats_cl_fin(const float* __restrict__ part, float* __restrict__ stats,
                     int C, int M)
{
  const int c = threadIdx.x;
  if (c >= C) return;
  float s = 0.f, q = 0.f;
  for (int b = 0; b < NB_STATS; ++b) {
    s += part[b * 512 + c];
    q += part[b * 512 + 256 + c];
  }
  const float inv = 1.f / (float)M;
  const float mu = s * inv;
  const float var = q * inv - mu * mu;
  stats[c] = mu;
  stats[256 + c] = rsqrtf(var + 1e-3f);
}

// y = g*(x-mean)*invstd + b ; += res ; tanh ; all bf16 channel-last w/ strides
__global__ __launch_bounds__(256)
void bn_apply_cl(const bf16_t* __restrict__ x, int ldx, int xoff,
                 const bf16_t* __restrict__ res, int ldr, int roff,
                 bf16_t* __restrict__ out, int ldo, int ooff,
                 const float* __restrict__ g, const float* __restrict__ b,
                 const float* __restrict__ stats, int C, int cshift, int M,
                 int do_tanh)
{
  const long i = (long)blockIdx.x * 256 + threadIdx.x;
  if (i >= ((long)M << cshift)) return;
  const long m = i >> cshift;
  const int  c = (int)(i & (C - 1));
  float y = g[c] * ((float)x[m * ldx + xoff + c] - stats[c]) * stats[256 + c] + b[c];
  if (res != nullptr) y += (float)res[m * ldr + roff + c];
  if (do_tanh) y = tanhf(y);
  out[m * ldo + ooff + c] = (bf16_t)y;
}

// channel_reduction: out[m][c] = cat[m][2c] + cat[m][2c+1]   (cat rows 2C wide)
__global__ __launch_bounds__(256)
void pairsum_cl(const bf16_t* __restrict__ cat, bf16_t* __restrict__ out,
                int C, int cshift, int M)
{
  const long i = (long)blockIdx.x * 256 + threadIdx.x;
  if (i >= ((long)M << cshift)) return;
  const long m = i >> cshift;
  const int  c = (int)(i & (C - 1));
  const long base = m * (long)(2 * C) + 2 * c;
  out[i] = (bf16_t)((float)cat[base] + (float)cat[base + 1]);
}

__global__ __launch_bounds__(256)
void add_cl(bf16_t* __restrict__ dst, const bf16_t* __restrict__ src, long n)
{
  const long i = (long)blockIdx.x * 256 + threadIdx.x;
  if (i < n) dst[i] = (bf16_t)((float)dst[i] + (float)src[i]);
}

// copy contiguous [M][C] bf16 into a strided destination (cat lower half)
__global__ __launch_bounds__(256)
void copy_cl(const bf16_t* __restrict__ in, bf16_t* __restrict__ out,
             int ldo, int ooff, int C, int cshift, int M)
{
  const long i = (long)blockIdx.x * 256 + threadIdx.x;
  if (i >= ((long)M << cshift)) return;
  const long m = i >> cshift;
  const int  c = (int)(i & (C - 1));
  out[m * ldo + ooff + c] = in[i];
}

// ---- f32 channel-first BN for the final 3-channel layer (output layout) ----
__global__ __launch_bounds__(256)
void bn_stats_f32(const float* __restrict__ x, float* __restrict__ stats,
                  int C, int M)
{
  __shared__ float ssum[256];
  __shared__ float ssq[256];
  const int c = blockIdx.x;
  const float* p = x + (long)c * M;
  float s = 0.f, q = 0.f;
  for (int i = threadIdx.x; i < M; i += 256) {
    const float v = p[i];
    s += v; q += v * v;
  }
  ssum[threadIdx.x] = s; ssq[threadIdx.x] = q;
  __syncthreads();
  for (int off = 128; off > 0; off >>= 1) {
    if ((int)threadIdx.x < off) {
      ssum[threadIdx.x] += ssum[threadIdx.x + off];
      ssq[threadIdx.x]  += ssq[threadIdx.x + off];
    }
    __syncthreads();
  }
  if (threadIdx.x == 0) {
    const float inv = 1.f / (float)M;
    const float mu = ssum[0] * inv;
    const float var = ssq[0] * inv - mu * mu;
    stats[c] = mu;
    stats[256 + c] = rsqrtf(var + 1e-3f);
  }
}

__global__ __launch_bounds__(256)
void bn_apply_f32(float* __restrict__ x, const float* __restrict__ g,
                  const float* __restrict__ b, const float* __restrict__ stats,
                  int C, int M)
{
  const long i = (long)blockIdx.x * 256 + threadIdx.x;
  if (i >= (long)C * M) return;
  const int c = (int)(i / M);
  x[i] = tanhf(g[c] * (x[i] - stats[c]) * stats[256 + c] + b[c]);
}

// ---------------------------------------------------------------------------
// Host-side parameter binding (insertion order vs JAX sorted-key order).
// ---------------------------------------------------------------------------
struct Blk { const float *w, *g, *b; };
struct Sbb { const float *w1, *bias1, *g1, *be1, *w2, *bias2, *g2, *be2; };
struct Params {
  const float* x;
  Blk ci, c1, c2[3], c3[3], c4[3];
  Sbb t4, t3, t2, t1;
  Blk m4, i4, m3, i3, m2, i2, m1, c5;
};

static void fill_insertion(void* const* din, Params& P) {
  int i = 0;
  auto nx  = [&]() { return (const float*)din[i++]; };
  auto blk = [&](Blk& B) { B.w = nx(); B.g = nx(); B.b = nx(); };
  auto sbb = [&](Sbb& S) { S.w1 = nx(); S.bias1 = nx(); S.g1 = nx(); S.be1 = nx();
                           S.w2 = nx(); S.bias2 = nx(); S.g2 = nx(); S.be2 = nx(); };
  P.x = nx();
  blk(P.ci); blk(P.c1);
  for (int j = 0; j < 3; ++j) blk(P.c2[j]);
  for (int j = 0; j < 3; ++j) blk(P.c3[j]);
  for (int j = 0; j < 3; ++j) blk(P.c4[j]);
  sbb(P.t4); blk(P.m4); blk(P.i4);
  sbb(P.t3); blk(P.m3); blk(P.i3);
  sbb(P.t2); blk(P.m2); blk(P.i2);
  sbb(P.t1); blk(P.m1); blk(P.c5);
}

static void fill_sorted(void* const* din, Params& P) {
  int i = 0;
  auto nx  = [&]() { return (const float*)din[i++]; };
  auto blk = [&](Blk& B) { B.b = nx(); B.g = nx(); B.w = nx(); };            // b,g,w
  auto sbb = [&](Sbb& S) { S.be1 = nx(); S.be2 = nx(); S.bias1 = nx(); S.bias2 = nx();
                           S.g1 = nx(); S.g2 = nx(); S.w1 = nx(); S.w2 = nx(); };
  blk(P.c1);
  for (int j = 0; j < 3; ++j) blk(P.c2[j]);
  for (int j = 0; j < 3; ++j) blk(P.c3[j]);
  for (int j = 0; j < 3; ++j) blk(P.c4[j]);
  blk(P.c5); blk(P.ci);
  blk(P.i2); blk(P.i3); blk(P.i4);
  blk(P.m1); blk(P.m2); blk(P.m3); blk(P.m4);
  sbb(P.t1); sbb(P.t2); sbb(P.t3); sbb(P.t4);
  P.x = nx();
}

static inline int ilog2i(int v) { int s = 0; while ((1 << s) < v) ++s; return s; }

// ---------------------------------------------------------------------------
extern "C" void kernel_launch(void* const* d_in, const int* in_sizes, int n_in,
                              void* d_out, int out_size, void* d_ws, size_t ws_size,
                              hipStream_t stream)
{
  (void)n_in; (void)out_size; (void)ws_size;
  Params P;
  if (in_sizes[0] == 3 * 64 * 64 * 64) fill_insertion(d_in, P);
  else                                 fill_sorted(d_in, P);

  const int M1 = 64 * 64 * 64, M2 = 32 * 32 * 32, M3 = 16 * 16 * 16, M4 = 7 * 8 * 8;

  // ---- workspace arena (byte bump allocator, deterministic) ----
  char* base = (char*)d_ws;
  auto alloc = [&](size_t bytes) {
    char* p = base;
    base += (bytes + 255) & ~(size_t)255;
    return p;
  };
  bf16_t* XIN = (bf16_t*)alloc(sizeof(bf16_t) * 32L * M1);   // padded input
  bf16_t* X1  = (bf16_t*)alloc(sizeof(bf16_t) * 32L * M1);
  bf16_t* X2  = (bf16_t*)alloc(sizeof(bf16_t) * 64L * M2);
  bf16_t* X3  = (bf16_t*)alloc(sizeof(bf16_t) * 128L * M3);
  bf16_t* X4  = (bf16_t*)alloc(sizeof(bf16_t) * 128L * M4);
  bf16_t* A   = (bf16_t*)alloc(sizeof(bf16_t) * 64L * M1);   // cat region
  bf16_t* Bt  = (bf16_t*)alloc(sizeof(bf16_t) * 32L * M1);
  bf16_t* Ct  = (bf16_t*)alloc(sizeof(bf16_t) * 32L * M1);
  bf16_t* WB  = (bf16_t*)alloc(sizeof(bf16_t) * 8L * 1024 * 1024);
  float*  ST  = (float*)alloc(sizeof(float) * 512);
  float*  PT  = (float*)alloc(sizeof(float) * NB_STATS * 512);
  bf16_t* wb_cur = WB;
  float* OUT = (float*)d_out;

  auto grid1 = [](long tot) { return (int)((tot + 255) / 256); };

  auto convk = [&](const bf16_t* in, const float* w, const float* bias,
                   bf16_t* out, float* out32, int ldo, int coff,
                   int Cin, int Cout, int Di, int Hi, int Wi,
                   int Do, int Ho, int Wo,
                   int sd, int sh, int sw, int pd, int ph, int pw, int ldil) {
    const int Cpad = (Cin < 32) ? 32 : Cin;            // all Cin are pow2
    const int Npad = ((Cout + 63) / 64) * 64;
    bf16_t* wb = wb_cur;
    const long wtot = (long)Npad * 27 * Cpad;
    wb_cur += wtot;
    repack_w<<<grid1(wtot), 256, 0, stream>>>(w, wb, Cout, Cin, Npad, Cpad);
    const int M = Do * Ho * Wo;
    const int mtiles = (M + 16 * MT - 1) / (16 * MT);
    dim3 grid((mtiles + WAVES_PER_BLOCK - 1) / WAVES_PER_BLOCK, Npad / 64, 1);
    conv3d_wmma_cl<<<grid, 256, 0, stream>>>(in, wb, bias, out, out32,
                                             Cpad, Cout, ldo, coff,
                                             Di, Hi, Wi, Do, Ho, Wo,
                                             sd, sh, sw, pd, ph, pw, ldil);
  };
  auto stats = [&](const bf16_t* x, int ldx, int xoff, int C, int M) {
    bn_stats_cl_part<<<NB_STATS, 256, 0, stream>>>(x, ldx, xoff, PT, C,
                                                   ilog2i(C), M);
    bn_stats_cl_fin<<<1, 256, 0, stream>>>(PT, ST, C, M);
  };
  auto bnap = [&](const bf16_t* x, int ldx, int xoff,
                  const bf16_t* res, int ldr, int roff,
                  bf16_t* out, int ldo, int ooff,
                  const float* g, const float* b, int C, int M) {
    bn_apply_cl<<<grid1((long)M * C), 256, 0, stream>>>(
        x, ldx, xoff, res, ldr, roff, out, ldo, ooff, g, b, ST, C, ilog2i(C), M, 1);
  };
  // conv -> BN(batch stats) -> tanh, contiguous channel-last in/out
  auto block = [&](const bf16_t* in, const Blk& B, bf16_t* out, int Cin, int Cout,
                   int Di, int Hi, int Wi, int Do, int Ho, int Wo,
                   int s, int pd, int ph, int pw) {
    convk(in, B.w, nullptr, out, nullptr, Cout, 0, Cin, Cout,
          Di, Hi, Wi, Do, Ho, Wo, s, s, s, pd, ph, pw, 1);
    stats(out, Cout, 0, Cout, Do * Ho * Wo);
    bnap(out, Cout, 0, nullptr, 0, 0, out, Cout, 0, B.g, B.b, Cout, Do * Ho * Wo);
  };

  // =========================== encoder ===========================
  import_x<<<grid1(32L * M1), 256, 0, stream>>>(P.x, XIN, M1);
  block(XIN, P.ci, Bt, 32, 32, 64,64,64, 64,64,64, 1, 1,1,1);       // conv_input
  block(Bt, P.c1, X1, 32, 32, 64,64,64, 64,64,64, 1, 1,1,1);        // conv1 -> x1
  block(X1, P.c2[0], Ct, 32, 64, 64,64,64, 32,32,32, 2, 1,1,1);     // conv2
  block(Ct, P.c2[1], Bt, 64, 64, 32,32,32, 32,32,32, 1, 1,1,1);
  block(Bt, P.c2[2], X2, 64, 64, 32,32,32, 32,32,32, 1, 1,1,1);     // -> x2
  block(X2, P.c3[0], Ct, 64, 128, 32,32,32, 16,16,16, 2, 1,1,1);    // conv3
  block(Ct, P.c3[1], Bt, 128, 128, 16,16,16, 16,16,16, 1, 1,1,1);
  block(Bt, P.c3[2], X3, 128, 128, 16,16,16, 16,16,16, 1, 1,1,1);   // -> x3
  block(X3, P.c4[0], Ct, 128, 128, 16,16,16, 7,8,8, 2, 0,1,1);      // conv4
  block(Ct, P.c4[1], Bt, 128, 128, 7,8,8, 7,8,8, 1, 1,1,1);
  block(Bt, P.c4[2], X4, 128, 128, 7,8,8, 7,8,8, 1, 1,1,1);         // -> x4

  // ======================= decoder (UR blocks) ====================
  // Invariant: "bottom" already sits in cat rows A[m][0..C); cat row = 2C wide.
  auto ur = [&](const bf16_t* lateral, int C, int Dd, int Hh, int Ww,
                const Sbb& T, const Blk& Mb) {
    const int M = Dd * Hh * Ww;
    // SparseBasicBlock on lateral
    convk(lateral, T.w1, T.bias1, Bt, nullptr, C, 0, C, C,
          Dd,Hh,Ww, Dd,Hh,Ww, 1,1,1, 1,1,1, 1);
    stats(Bt, C, 0, C, M);
    bnap(Bt, C, 0, nullptr, 0, 0, Bt, C, 0, T.g1, T.be1, C, M);
    convk(Bt, T.w2, T.bias2, Ct, nullptr, C, 0, C, C,
          Dd,Hh,Ww, Dd,Hh,Ww, 1,1,1, 1,1,1, 1);
    stats(Ct, C, 0, C, M);
    bnap(Ct, C, 0, lateral, C, 0, A, 2 * C, C, T.g2, T.be2, C, M); // x_trans->cat hi
    // x_m = block(cat): 2C -> C
    convk(A, Mb.w, nullptr, Ct, nullptr, C, 0, 2 * C, C,
          Dd,Hh,Ww, Dd,Hh,Ww, 1,1,1, 1,1,1, 1);
    stats(Ct, C, 0, C, M);
    bnap(Ct, C, 0, nullptr, 0, 0, Ct, C, 0, Mb.g, Mb.b, C, M);
    // red = pair-sum over cat channels; z = x_m + red (in Ct)
    pairsum_cl<<<grid1((long)M * C), 256, 0, stream>>>(A, Bt, C, ilog2i(C), M);
    add_cl<<<grid1((long)M * C), 256, 0, stream>>>(Ct, Bt, (long)M * C);
  };

  // level 4: lateral = bottom = x4
  copy_cl<<<grid1(128L * M4), 256, 0, stream>>>(X4, A, 256, 0, 128, 7, M4);
  ur(X4, 128, 7,8,8, P.t4, P.m4);
  // inv4: transposed conv 128->128, (7,8,8)->(16,16,16), pad_lo (2,1,1)
  // -> straight into level-3 cat lower half (rows 256 wide)
  convk(Ct, P.i4.w, nullptr, A, nullptr, 256, 0, 128, 128,
        7,8,8, 16,16,16, 1,1,1, 2,1,1, 2);
  stats(A, 256, 0, 128, M3);
  bnap(A, 256, 0, nullptr, 0, 0, A, 256, 0, P.i4.g, P.i4.b, 128, M3);  // up4

  // level 3
  ur(X3, 128, 16,16,16, P.t3, P.m3);
  convk(Ct, P.i3.w, nullptr, A, nullptr, 128, 0, 128, 64,
        16,16,16, 32,32,32, 1,1,1, 1,1,1, 2);
  stats(A, 128, 0, 64, M2);
  bnap(A, 128, 0, nullptr, 0, 0, A, 128, 0, P.i3.g, P.i3.b, 64, M2);   // up3

  // level 2
  ur(X2, 64, 32,32,32, P.t2, P.m2);
  convk(Ct, P.i2.w, nullptr, A, nullptr, 64, 0, 64, 32,
        32,32,32, 64,64,64, 1,1,1, 1,1,1, 2);
  stats(A, 64, 0, 32, M1);
  bnap(A, 64, 0, nullptr, 0, 0, A, 64, 0, P.i2.g, P.i2.b, 32, M1);     // up2

  // level 1: final up block uses a normal conv block (conv5: 32 -> 3),
  // stored fp32 channel-first directly into d_out.
  ur(X1, 32, 64,64,64, P.t1, P.m1);
  convk(Ct, P.c5.w, nullptr, nullptr, OUT, 0, 0, 32, 3,
        64,64,64, 64,64,64, 1,1,1, 1,1,1, 1);
  bn_stats_f32<<<3, 256, 0, stream>>>(OUT, ST, 3, M1);
  bn_apply_f32<<<grid1(3L * M1), 256, 0, stream>>>(OUT, P.c5.g, P.c5.b, ST, 3, M1);
}
